// SelfAttentionBlock_34050500722772
// MI455X (gfx1250) — matrix operand order repaired and checked by reading, not compiled, not run on hardware
//
#include <hip/hip_runtime.h>
#include <hip/hip_bf16.h>

// Problem constants (from reference)
#define NB     8
#define T_SEQ  2048
#define C_IN   512
#define BT     (NB * T_SEQ)        // 16384 rows
#define OUTC   1024                // concat(x, attn_out)

typedef __attribute__((ext_vector_type(16))) __bf16       v16bf;
typedef __attribute__((ext_vector_type(8)))  float        v8f;
typedef __attribute__((ext_vector_type(8)))  unsigned int v8u;

// fp32 -> bf16 with round-to-nearest-even
static __device__ __forceinline__ unsigned short f2bf(float f) {
  unsigned u = __builtin_bit_cast(unsigned, f);
  u += 0x7fffu + ((u >> 16) & 1u);
  return (unsigned short)(u >> 16);
}
static __device__ __forceinline__ unsigned packbf(float lo, float hi) {
  return (unsigned)f2bf(lo) | ((unsigned)f2bf(hi) << 16);
}

static __device__ __forceinline__ v8f wmma_bf16(v8u a, v8u b, v8f c) {
  return __builtin_amdgcn_wmma_f32_16x16x32_bf16(
      false, __builtin_bit_cast(v16bf, a),
      false, __builtin_bit_cast(v16bf, b),
      (short)0, c, false, false);
}

// ---- fragment loaders -------------------------------------------------------
// 16-bit A fragment (16x32): lane holds two contiguous 16-byte runs.
static __device__ __forceinline__ v8u load_frag_a(const unsigned short* row,
                                                  int half) {
  const uint4 lo = *reinterpret_cast<const uint4*>(row + 8 * half);
  const uint4 hi = *reinterpret_cast<const uint4*>(row + 16 + 8 * half);
  v8u a;
  a[0] = lo.x; a[1] = lo.y; a[2] = lo.z; a[3] = lo.w;
  a[4] = hi.x; a[5] = hi.y; a[6] = hi.z; a[7] = hi.w;
  return a;
}
// 16-bit B fragment (32x16): lane holds two contiguous 16-byte runs.
static __device__ __forceinline__ v8u load_frag_b(const unsigned short* col,
                                                  int half) {
  const uint4 lo = *reinterpret_cast<const uint4*>(col + 8 * half);
  const uint4 hi = *reinterpret_cast<const uint4*>(col + 16 + 8 * half);
  v8u b;
  b[0] = lo.x; b[1] = lo.y; b[2] = lo.z; b[3] = lo.w;
  b[4] = hi.x; b[5] = hi.y; b[6] = hi.z; b[7] = hi.w;
  return b;
}

// ---------------------------------------------------------------------------
// Kernel 1: out[:, :, 0:512] = x  (identity half of the concat)
// ---------------------------------------------------------------------------
__global__ __launch_bounds__(256) void copyx_kernel(const float4* __restrict__ x4,
                                                    float* __restrict__ out) {
  const size_t i   = (size_t)blockIdx.x * blockDim.x + threadIdx.x; // < BT*128
  const size_t row = i >> 7;
  const size_t c4  = i & 127;
  float4 v = x4[i];
  *reinterpret_cast<float4*>(out + row * OUTC + c4 * 4) = v;
}

// ---------------------------------------------------------------------------
// Kernel 2: prepack x -> bf16 (row-major, [BT,512]); 8 elements per thread
// ---------------------------------------------------------------------------
__global__ __launch_bounds__(256) void packx_kernel(const float4* __restrict__ x4,
                                                    unsigned short* __restrict__ xbf) {
  const size_t i = (size_t)blockIdx.x * blockDim.x + threadIdx.x; // < BT*512/8
  const float4 a = x4[2 * i];
  const float4 b = x4[2 * i + 1];
  uint4 p;
  p.x = packbf(a.x, a.y);
  p.y = packbf(a.z, a.w);
  p.z = packbf(b.x, b.y);
  p.w = packbf(b.z, b.w);
  *reinterpret_cast<uint4*>(xbf + 8 * i) = p;
}

// ---------------------------------------------------------------------------
// Kernel 3: prepack W (fp32 [512(c),512(n)]) -> Wt (bf16 [512(n),512(c)]).
// ---------------------------------------------------------------------------
__global__ __launch_bounds__(256) void packw_kernel(
    const float* __restrict__ Wq, const float* __restrict__ Wk,
    const float* __restrict__ Wv, unsigned short* __restrict__ WtAll) {
  const float* W = (blockIdx.z == 0) ? Wq : (blockIdx.z == 1) ? Wk : Wv;
  unsigned short* Wt = WtAll + (size_t)blockIdx.z * C_IN * C_IN;
  const size_t idx = (size_t)blockIdx.x * blockDim.x + threadIdx.x; // < 512*64
  const int n  = (int)(idx & 511);
  const int c0 = (int)(idx >> 9) * 8;
  uint4 p;
  p.x = packbf(W[(size_t)(c0 + 0) * C_IN + n], W[(size_t)(c0 + 1) * C_IN + n]);
  p.y = packbf(W[(size_t)(c0 + 2) * C_IN + n], W[(size_t)(c0 + 3) * C_IN + n]);
  p.z = packbf(W[(size_t)(c0 + 4) * C_IN + n], W[(size_t)(c0 + 5) * C_IN + n]);
  p.w = packbf(W[(size_t)(c0 + 6) * C_IN + n], W[(size_t)(c0 + 7) * C_IN + n]);
  *reinterpret_cast<uint4*>(Wt + (size_t)n * C_IN + c0) = p;
}

// ---------------------------------------------------------------------------
// Kernel 4: q/k/v projections, pure bf16 WMMA GEMM.
//   mat 0 -> q (bf16, [BT,512]); mat 1 -> k; mat 2 -> vT (bf16, [512, BT]).
// ---------------------------------------------------------------------------
__global__ __launch_bounds__(256) void proj_kernel(
    const unsigned short* __restrict__ xbf,
    const unsigned short* __restrict__ WtAll,
    const float* __restrict__ bq, const float* __restrict__ bk,
    const float* __restrict__ bv,
    unsigned short* __restrict__ qws, unsigned short* __restrict__ kws,
    unsigned short* __restrict__ vT) {
  const int lane = threadIdx.x & 31;
  const int wave = threadIdx.x >> 5;
  const int half = lane >> 4;
  const int ml   = lane & 15;

  const int mat = blockIdx.z;
  const unsigned short* Wt = WtAll + (size_t)mat * C_IN * C_IN;
  const float* bias = (mat == 0) ? bq : (mat == 1) ? bk : bv;

  const int rowBase = blockIdx.x * 64 + (wave >> 1) * 16;
  const int colBase = blockIdx.y * 128 + (wave & 1) * 64;

  v8f acc[4] = {};
  const unsigned short* arow = xbf + (size_t)(rowBase + ml) * C_IN;

  for (int c0 = 0; c0 < C_IN; c0 += 32) {
    const v8u a = load_frag_a(arow + c0, half);
#pragma unroll
    for (int t = 0; t < 4; ++t) {
      const int ncol = colBase + t * 16 + ml;
      const v8u bfrag = load_frag_b(Wt + (size_t)ncol * C_IN + c0, half);
      acc[t] = wmma_bf16(a, bfrag, acc[t]);
    }
  }

  // Bias + store
#pragma unroll
  for (int t = 0; t < 4; ++t) {
    const int ncol = colBase + t * 16 + ml;
    const float bb = bias[ncol];
    if (mat < 2) {
      unsigned short* dst = (mat == 0) ? qws : kws;
#pragma unroll
      for (int r = 0; r < 8; ++r) {
        const int m = rowBase + 8 * half + r;
        dst[(size_t)m * C_IN + ncol] = f2bf(acc[t][r] + bb);
      }
    } else {
      // transposed V store: rows r contiguous for fixed lane -> one b128
      uint4 vv = make_uint4(packbf(acc[t][0] + bb, acc[t][1] + bb),
                            packbf(acc[t][2] + bb, acc[t][3] + bb),
                            packbf(acc[t][4] + bb, acc[t][5] + bb),
                            packbf(acc[t][6] + bb, acc[t][7] + bb));
      *reinterpret_cast<uint4*>(vT + (size_t)ncol * BT + rowBase + 8 * half) = vv;
    }
  }
}

// ---------------------------------------------------------------------------
// Kernel 5: causal flash attention, 64-key chunks.
// Grid: (T/64, B). Block: 256 (8 waves). Wave = (qw = wave/2, vw = wave%2):
//   16 queries x 256 v-dims per wave (16 f32 WMMA accumulator tiles).
// Per chunk: 4 score tiles (16x16), one softmax pass, P = 16x64 bf16 in LDS,
// PV uses two A fragments x 16 B fragments.
// ---------------------------------------------------------------------------
__global__ __launch_bounds__(256) void attn_kernel(
    const unsigned short* __restrict__ qws,
    const unsigned short* __restrict__ kws,
    const unsigned short* __restrict__ vT,
    float* __restrict__ out) {
  __shared__ __align__(16) unsigned short p_lds[8][16 * 64];

  const int lane = threadIdx.x & 31;
  const int wave = threadIdx.x >> 5;
  const int half = lane >> 4;
  const int ml   = lane & 15;
  const int qw   = wave >> 1;
  const int vw   = wave & 1;
  const int b    = blockIdx.y;
  const int q0   = blockIdx.x * 64 + qw * 16;

  const unsigned short* qrow = qws + ((size_t)b * T_SEQ + q0 + ml) * C_IN;
  const unsigned short* kb   = kws + (size_t)b * T_SEQ * C_IN;

  v8f acc[16] = {};
  float rmax[8], rsum[8];
#pragma unroll
  for (int r = 0; r < 8; ++r) { rmax[r] = -3.0e38f; rsum[r] = 0.0f; }

  const float sc = 0.04419417382415922f; // 1/sqrt(512)
  const int kend = q0 + 15;              // last key needed (causal)

  for (int j0 = 0; j0 <= kend; j0 += 64) {
    // ---- S = Q @ K^T over full 512-d, four 16-key tiles ----
    v8f s[4] = {};
    const unsigned short* krow[4];
#pragma unroll
    for (int i = 0; i < 4; ++i)
      krow[i] = kb + (size_t)(j0 + 16 * i + ml) * C_IN;

    for (int c0 = 0; c0 < C_IN; c0 += 32) {
      const v8u a = load_frag_a(qrow + c0, half);
#pragma unroll
      for (int i = 0; i < 4; ++i) {
        const v8u kf = load_frag_b(krow[i] + c0, half);
        s[i] = wmma_bf16(a, kf, s[i]);
      }
    }

    // ---- scale + causal mask + online softmax (16-lane row reductions) ----
    float corr[8];
#pragma unroll
    for (int r = 0; r < 8; ++r) {
      const int m = q0 + 8 * half + r;
      float e[4];
#pragma unroll
      for (int i = 0; i < 4; ++i)
        e[i] = (j0 + 16 * i + ml <= m) ? s[i][r] * sc : -3.0e38f;
      float cm = fmaxf(fmaxf(e[0], e[1]), fmaxf(e[2], e[3]));
      cm = fmaxf(cm, __shfl_xor(cm, 1, 32));
      cm = fmaxf(cm, __shfl_xor(cm, 2, 32));
      cm = fmaxf(cm, __shfl_xor(cm, 4, 32));
      cm = fmaxf(cm, __shfl_xor(cm, 8, 32));
      const float nm = fmaxf(rmax[r], cm);
      float p[4];
#pragma unroll
      for (int i = 0; i < 4; ++i) p[i] = __expf(e[i] - nm);
      float cs = (p[0] + p[1]) + (p[2] + p[3]);
      cs += __shfl_xor(cs, 1, 32);
      cs += __shfl_xor(cs, 2, 32);
      cs += __shfl_xor(cs, 4, 32);
      cs += __shfl_xor(cs, 8, 32);
      const float cr = __expf(rmax[r] - nm);
      rsum[r] = rsum[r] * cr + cs;
      rmax[r] = nm;
      corr[r] = cr;
      // stage P (C-layout -> row-major bf16 16x64 tile) for A-fragment reads
      unsigned short* prow = &p_lds[wave][(8 * half + r) * 64];
#pragma unroll
      for (int i = 0; i < 4; ++i) prow[16 * i + ml] = f2bf(p[i]);
    }

    // wave-private LDS region: DS-counter drain only, no block barrier
    asm volatile("s_wait_dscnt 0x0" ::: "memory");

    // ---- read P back as two A fragments (keys 0-31, 32-63 of the chunk) ----
    const v8u pa0 = load_frag_a(&p_lds[wave][ml * 64], half);
    const v8u pa1 = load_frag_a(&p_lds[wave][ml * 64 + 32], half);

    // ---- acc = acc*corr + P @ V ----
#pragma unroll
    for (int tt = 0; tt < 16; ++tt) {
      const int n = vw * 256 + tt * 16 + ml;
      const unsigned short* vp = vT + (size_t)n * BT + (size_t)b * T_SEQ + j0;
      const v8u bv0 = load_frag_b(vp, half);
      const v8u bv1 = load_frag_b(vp + 32, half);
      v8f c = acc[tt];
#pragma unroll
      for (int r = 0; r < 8; ++r) c[r] *= corr[r];
      c = wmma_bf16(pa0, bv0, c);
      acc[tt] = wmma_bf16(pa1, bv1, c);
    }
  }

  // ---- epilogue: out[:, :, 512:1024] = acc / rowsum ----
  float inv[8];
#pragma unroll
  for (int r = 0; r < 8; ++r) inv[r] = 1.0f / rsum[r];
#pragma unroll
  for (int tt = 0; tt < 16; ++tt) {
    const int n = vw * 256 + tt * 16 + ml;
#pragma unroll
    for (int r = 0; r < 8; ++r) {
      const int m = q0 + 8 * half + r;
      out[((size_t)b * T_SEQ + m) * OUTC + 512 + n] = acc[tt][r] * inv[r];
    }
  }
}

// ---------------------------------------------------------------------------
extern "C" void kernel_launch(void* const* d_in, const int* in_sizes, int n_in,
                              void* d_out, int out_size, void* d_ws, size_t ws_size,
                              hipStream_t stream) {
  const float* x  = (const float*)d_in[0];
  const float* Wq = (const float*)d_in[1];
  const float* bq = (const float*)d_in[2];
  const float* Wk = (const float*)d_in[3];
  const float* bk = (const float*)d_in[4];
  const float* Wv = (const float*)d_in[5];
  const float* bv = (const float*)d_in[6];
  float* out = (float*)d_out;

  // Workspace (bf16): q, k [BT,512]; vT [512,BT]; xbf [BT,512]; Wt 3x[512,512]
  unsigned short* qws = (unsigned short*)d_ws;
  unsigned short* kws = qws + (size_t)BT * C_IN;
  unsigned short* vT  = kws + (size_t)BT * C_IN;
  unsigned short* xbf = vT  + (size_t)C_IN * BT;
  unsigned short* Wt  = xbf + (size_t)BT * C_IN;   // 3 * 512*512

  // out[:, :, :512] = x
  copyx_kernel<<<dim3((BT * 128) / 256), dim3(256), 0, stream>>>(
      (const float4*)x, out);

  // prepack activations and weights to bf16 (weights transposed)
  packx_kernel<<<dim3((BT * C_IN / 8) / 256), dim3(256), 0, stream>>>(
      (const float4*)x, xbf);
  packw_kernel<<<dim3((C_IN * (C_IN / 8)) / 256, 1, 3), dim3(256), 0, stream>>>(
      Wq, Wk, Wv, Wt);

  // q/k/v projections (z = matrix select)
  proj_kernel<<<dim3(BT / 64, C_IN / 128, 3), dim3(256), 0, stream>>>(
      xbf, Wt, bq, bk, bv, qws, kws, vT);

  // causal flash attention -> out[:, :, 512:]
  attn_kernel<<<dim3(T_SEQ / 64, NB), dim3(256), 0, stream>>>(qws, kws, vT, out);
}
